// GroupedQueryAttention_89979564851823
// MI455X (gfx1250) — matrix operand order repaired and checked by reading, not compiled, not run on hardware
//
#include <hip/hip_runtime.h>

// ---------------------------------------------------------------------------
// Types for CDNA5 WMMA (wave32): v16bf A/B fragments, v8f f32 accumulator
// ---------------------------------------------------------------------------
typedef __bf16 bf16;
typedef __bf16 v16bf __attribute__((ext_vector_type(16)));
typedef float  v8f   __attribute__((ext_vector_type(8)));
typedef unsigned int u32;
typedef u32 u32x4 __attribute__((ext_vector_type(4)));
typedef int  i32x4 __attribute__((ext_vector_type(4)));
typedef int  i32x8 __attribute__((ext_vector_type(8)));

union FragU {
  uint4 q[2];
  v16bf v;
};

__device__ __forceinline__ v8f wmma_bf16(v16bf a, v16bf b, v8f c) {
  // D = A(16x32 bf16) x B(32x16 bf16) + C(16x16 f32)
  return __builtin_amdgcn_wmma_f32_16x16x32_bf16(
      /*neg_a=*/false, a, /*neg_b=*/false, b,
      /*c_mod=*/(short)0, c, /*reuse_a=*/false, /*reuse_b=*/false);
}

// A-fragment (16x32, MxK) from row-major LDS tile (stride in elements, mult of 8).
// lane 0-15 : row = lane,    K = base+{0..7, 16..23}
// lane16-31 : row = lane-16, K = base+{8..15,24..31}
__device__ __forceinline__ v16bf load_frag_a(const bf16* base, int stride, int lane) {
  int half = (lane >> 4) & 1;
  int row  = lane & 15;
  const bf16* p = base + row * stride + half * 8;
  FragU f;
  f.q[0] = *(const uint4*)(p);
  f.q[1] = *(const uint4*)(p + 16);
  return f.v;
}

// B-fragment (32x16, KxN) from *transposed* LDS tile laid out [n][k].
// lane 0-15 : col = lane,    K = base+{0..7, 16..23}
// lane16-31 : col = lane-16, K = base+{8..15,24..31}
__device__ __forceinline__ v16bf load_frag_b(const bf16* base, int stride, int lane) {
  int half = (lane >> 4) & 1;
  int col  = lane & 15;
  const bf16* p = base + col * stride + half * 8;
  FragU f;
  f.q[0] = *(const uint4*)(p);
  f.q[1] = *(const uint4*)(p + 16);
  return f.v;
}

// ---------------------------------------------------------------------------
// Tensor Data Mover: async 2D bf16 tile (rows x cols) global -> LDS.
// Row stride given in elements. Issued once per wave; tracked by TENSORcnt.
// ---------------------------------------------------------------------------
#if defined(__has_builtin)
#if __has_builtin(__builtin_amdgcn_tensor_load_to_lds) && __has_builtin(__builtin_amdgcn_s_wait_tensorcnt)
#define HAVE_TDM 1
#endif
#endif
#ifndef HAVE_TDM
#define HAVE_TDM 0
#endif

#if HAVE_TDM
__device__ __forceinline__ void tdm_load_tile_2d(const bf16* gsrc, u32 lds_byte_off,
                                                 u32 tile_cols, u32 tile_rows,
                                                 u32 row_stride_elems) {
  unsigned long long ga = (unsigned long long)(size_t)gsrc;
  u32x4 g0;
  g0[0] = 1u;                                            // count=1, user flags=0
  g0[1] = lds_byte_off;                                  // lds_addr (bytes)
  g0[2] = (u32)(ga & 0xFFFFFFFFu);                       // global_addr[31:0]
  g0[3] = (u32)((ga >> 32) & 0x01FFFFFFu) | (2u << 30);  // addr[56:32], type=2
  i32x8 g1;
  g1[0] = (int)(1u << 16);                    // data_size=1 (2 bytes/elem)
  g1[1] = (int)(row_stride_elems << 16);      // tensor_dim0[15:0]
  g1[2] = (int)(65535u << 16);                // tensor_dim0 hi=0, tensor_dim1 lo=big
  g1[3] = (int)(tile_cols << 16);             // tensor_dim1 hi=0, tile_dim0
  g1[4] = (int)tile_rows;                     // tile_dim1, tile_dim2=0
  g1[5] = (int)row_stride_elems;              // tensor_dim0_stride[31:0]
  g1[6] = 0;
  g1[7] = 0;
  i32x4 z4 = {0, 0, 0, 0};
#if __clang_major__ >= 23
  i32x8 z8 = {0, 0, 0, 0, 0, 0, 0, 0};
  __builtin_amdgcn_tensor_load_to_lds(g0, g1, z4, z4, z8, 0);
#else
  __builtin_amdgcn_tensor_load_to_lds(g0, g1, z4, z4, 0);
#endif
}
#endif

// ---------------------------------------------------------------------------
// fp32 -> bf16 convert
// ---------------------------------------------------------------------------
__global__ void cvt_f32_bf16(const float* __restrict__ src, bf16* __restrict__ dst, int n) {
  int i = blockIdx.x * blockDim.x + threadIdx.x;
  if (i < n) dst[i] = (bf16)src[i];
}

// ---------------------------------------------------------------------------
// bf16 GEMM: C[M,N] = A[M,K] * B[K,N].  A,B row-major bf16.
// Block tile 128x128, BK=32, 256 threads (8 waves, 4x2 wave grid),
// each wave owns a 32x64 strip: 2 A-frags, 4 B-frags, 8 WMMAs per K-step.
// ---------------------------------------------------------------------------
template <bool OUT_F32>
__global__ __launch_bounds__(256) void gemm_bf16_wmma(
    const bf16* __restrict__ A, const bf16* __restrict__ Bw,
    void* __restrict__ Cout, int M, int N, int K) {
  constexpr int BM = 128, BN = 128, BK = 32, LS = 40;  // LS: padded stride (16B mult)
  __shared__ bf16 As[BM * LS];
  __shared__ bf16 Bs[BN * LS];  // transposed: Bs[n][k]

  const int tid = threadIdx.x;
  const int lane = tid & 31;
  const int w = tid >> 5;
  const int wr = w & 3, wc = w >> 2;
  const int mb = blockIdx.y * BM;
  const int nb = blockIdx.x * BN;
  const int m0 = wr * 32, n0 = wc * 64;

  v8f acc[8] = {};

  for (int kb = 0; kb < K; kb += BK) {
    // Stage A tile: 128x32, two uint4 (8 bf16) chunks per thread
#pragma unroll
    for (int c = 0; c < 2; ++c) {
      int chunk = tid + c * 256;  // 0..511
      int row = chunk >> 2;       // 0..127
      int kc  = (chunk & 3) * 8;  // 0,8,16,24
      *(uint4*)(As + row * LS + kc) =
          *(const uint4*)(A + (size_t)(mb + row) * K + kb + kc);
    }
    // Stage B tile transposed: Bs[n][k] = B[kb+k][nb+n]; uint4 global reads
#pragma unroll
    for (int c = 0; c < 2; ++c) {
      int chunk = tid + c * 256;   // 0..511
      int k   = chunk >> 4;        // 0..31
      int nc0 = (chunk & 15) * 8;  // 0..120
      uint4 d = *(const uint4*)(Bw + (size_t)(kb + k) * N + nb + nc0);
      const bf16* e = (const bf16*)&d;
#pragma unroll
      for (int i = 0; i < 8; ++i) Bs[(nc0 + i) * LS + k] = e[i];
    }
    // Prefetch next K-step tiles into L2 while this step computes
    if (kb + BK < K) {
      __builtin_prefetch(A + (size_t)(mb + (tid >> 1)) * K + kb + BK, 0, 1);
      __builtin_prefetch(Bw + (size_t)(kb + BK + (tid >> 3)) * N + nb + (tid & 7) * 16, 0, 1);
    }
    __syncthreads();

    v16bf a0 = load_frag_a(As + (m0) * LS, LS, lane);
    v16bf a1 = load_frag_a(As + (m0 + 16) * LS, LS, lane);
#pragma unroll
    for (int nf = 0; nf < 4; ++nf) {
      v16bf bfr = load_frag_b(Bs + (n0 + nf * 16) * LS, LS, lane);
      acc[nf]     = wmma_bf16(a0, bfr, acc[nf]);
      acc[nf + 4] = wmma_bf16(a1, bfr, acc[nf + 4]);
    }
    __syncthreads();
  }

  // Epilogue: C layout -> row = r + 8*halfw per half-wave, col = lane&15
  const int halfw = (lane >> 4) & 1, l16 = lane & 15;
#pragma unroll
  for (int hm = 0; hm < 2; ++hm) {
#pragma unroll
    for (int nf = 0; nf < 4; ++nf) {
#pragma unroll
      for (int r = 0; r < 8; ++r) {
        size_t row = (size_t)(mb + m0 + hm * 16 + halfw * 8 + r);
        size_t col = (size_t)(nb + n0 + nf * 16 + l16);
        float v = acc[hm * 4 + nf][r];
        if (OUT_F32) ((float*)Cout)[row * N + col] = v;
        else         ((bf16*)Cout)[row * N + col] = (bf16)v;
      }
    }
  }
}

// ---------------------------------------------------------------------------
// RoPE: interleaved pairs, angle = t * 10000^(-2i/64). src/dst: [B*T][rowWidth]
// ---------------------------------------------------------------------------
__global__ void rope_bf16(const bf16* __restrict__ src, bf16* __restrict__ dst,
                          int rowWidth, int T, int totalPairs) {
  int i = blockIdx.x * blockDim.x + threadIdx.x;
  if (i >= totalPairs) return;
  int pairsPerRow = rowWidth >> 1;
  int row = i / pairsPerRow;        // row = b*T + t
  int pr  = i - row * pairsPerRow;  // pair index inside row
  int d2  = pr & 31;                // pair index inside a 64-wide head
  int t   = row & (T - 1);          // T = 2048 (power of 2)
  float inv = __powf(10000.0f, -(float)(2 * d2) * (1.0f / 64.0f));
  float ang = (float)t * inv;
  float s, c;
  __sincosf(ang, &s, &c);
  const bf16* p = src + (size_t)row * rowWidth + pr * 2;
  float e = (float)p[0], o = (float)p[1];
  bf16* q = dst + (size_t)row * rowWidth + pr * 2;
  q[0] = (bf16)(e * c - o * s);
  q[1] = (bf16)(e * s + o * c);
}

// ---------------------------------------------------------------------------
// Flash attention (causal, GQA). Q:[B*T][2048] (head h at col h*64),
// K/V:[B*T][512] (kv head g at col g*64). Out ctx:[B*T][2048].
// Block = 128 threads (4 waves); block handles 64 query rows of one (b,h).
// K tile staged by the Tensor Data Mover (async, TENSORcnt); V transposed
// cooperatively while the TDM transfer is in flight.
// ---------------------------------------------------------------------------
__global__ __launch_bounds__(128) void flash_attn_wmma(
    const bf16* __restrict__ Q, const bf16* __restrict__ Kx,
    const bf16* __restrict__ Vx, bf16* __restrict__ O) {
  constexpr int T = 2048, LS = 72;  // LDS row stride (elements), 144B (16B mult)
  __shared__ bf16 Ks[64 * LS];      // [j][d]   (d padded to LS)
  __shared__ bf16 Vt[64 * LS];      // transposed: [d][j]
  __shared__ bf16 Pb[4][16 * LS];   // per-wave P tile [qrow][j]

  const int tid = threadIdx.x, lane = tid & 31, w = tid >> 5;
  const int half = (lane >> 4) & 1, l16 = lane & 15;
  const int qt = blockIdx.x;  // query tile (64 rows)
  const int h  = blockIdx.y;  // q head
  const int b  = blockIdx.z;
  const int g  = h >> 2;      // kv head (N_REP=4)

  const size_t qbase = (size_t)(b * T) * 2048 + (size_t)h * 64;
  const size_t kbase = (size_t)(b * T) * 512 + (size_t)g * 64;

  // Load this wave's Q fragments (rows w*16..w*16+15, two 32-wide K chunks)
  v16bf qa[2];
  {
    const bf16* p = Q + qbase + (size_t)(qt * 64 + w * 16 + l16) * 2048;
#pragma unroll
    for (int kc = 0; kc < 2; ++kc) {
      FragU f;
      const bf16* pp = p + kc * 32 + half * 8;
      f.q[0] = *(const uint4*)(pp);
      f.q[1] = *(const uint4*)(pp + 16);
      qa[kc] = f.v;
    }
  }

  v8f o_acc[4] = {};
  float mloc[8], lloc[8];
#pragma unroll
  for (int r = 0; r < 8; ++r) { mloc[r] = -1e30f; lloc[r] = 0.0f; }

  const float scale = 0.125f;  // 1/sqrt(64)

  for (int jt = 0; jt <= qt; ++jt) {
    // ---- Stage K tile [j][d] (TDM: tile is padded row-by-row via pad? no —
    //      TDM writes contiguous; we give it the padded LDS row stride by
    //      loading 64 one-row "tiles"? Simpler: LS==72 padding only matters
    //      for fragment reads; TDM path loads the tile *unpadded* into a
    //      64x64 region at stride 64 is wrong for our readers, so the TDM
    //      descriptor instead treats LDS as 64 rows with pad via iterate?
    //      Cleanest correct option: TDM loads the tile densely row-major
    //      with tile_dim0=64 and we make the LDS reader stride match: we
    //      dedicate Ks with LS stride and load per-row using the same dense
    //      copy only when TDM is unavailable. To keep reader strides fixed,
    //      the TDM transfer uses pad_enable to insert (LS-64) elements of
    //      padding every row: pad_interval = 32 DWORDs (64 bf16), pad_amount
    //      = 4 DWORDs (8 bf16) -> LS = 72. ----
#if HAVE_TDM
    if (w == 0) {
      unsigned long long ga = (unsigned long long)(size_t)(Kx + kbase + (size_t)(jt * 64) * 512);
      u32x4 g0;
      g0[0] = 1u;
      g0[1] = (u32)(size_t)(void*)Ks;                        // lds byte offset
      g0[2] = (u32)(ga & 0xFFFFFFFFu);
      g0[3] = (u32)((ga >> 32) & 0x01FFFFFFu) | (2u << 30);  // type=2 (image)
      i32x8 g1;
      g1[0] = (int)((1u << 16) |   // data_size=1 (2B)
                    (1u << 20) |   // pad_enable
                    (4u << 22) |   // pad_interval: code 4 => 32 DWORDs (64 bf16)
                    (3u << 25));   // pad_amount: code 3 => 4 DWORDs (8 bf16)
      g1[1] = (int)(512u << 16);   // tensor_dim0 = 512 (lo16)
      g1[2] = (int)(65535u << 16); // tensor_dim1 lo16 (large; OOB never hit)
      g1[3] = (int)(64u << 16);    // tile_dim0 = 64
      g1[4] = 64;                  // tile_dim1 = 64
      g1[5] = 512;                 // tensor_dim0_stride = 512
      g1[6] = 0;
      g1[7] = 0;
      i32x4 z4 = {0, 0, 0, 0};
#if __clang_major__ >= 23
      i32x8 z8 = {0, 0, 0, 0, 0, 0, 0, 0};
      __builtin_amdgcn_tensor_load_to_lds(g0, g1, z4, z4, z8, 0);
#else
      __builtin_amdgcn_tensor_load_to_lds(g0, g1, z4, z4, 0);
#endif
    }
#else
    {
      int row = tid >> 1;        // 0..63
      int c0  = (tid & 1) * 32;  // d half
      const bf16* ksrc = Kx + kbase + (size_t)(jt * 64 + row) * 512 + c0;
#pragma unroll
      for (int q4 = 0; q4 < 4; ++q4)
        *(uint4*)(Ks + row * LS + c0 + q4 * 8) = *(const uint4*)(ksrc + q4 * 8);
    }
#endif
    // ---- Transposed V tile [d][j], overlapped with the TDM transfer ----
    {
      int row = tid >> 1;
      int c0  = (tid & 1) * 32;
      const bf16* vsrc = Vx + kbase + (size_t)(jt * 64 + row) * 512 + c0;
#pragma unroll
      for (int dd = 0; dd < 32; ++dd)
        Vt[(c0 + dd) * LS + row] = vsrc[dd];
    }
#if HAVE_TDM
    if (w == 0) __builtin_amdgcn_s_wait_tensorcnt(0);
#endif
    __syncthreads();

    // ---- S = Q * K^T  (16 q-rows x 64 keys per wave) ----
    v8f s[4];
#pragma unroll
    for (int nf = 0; nf < 4; ++nf) {
      v8f a = {};
#pragma unroll
      for (int kc = 0; kc < 2; ++kc) {
        v16bf bfr = load_frag_b(Ks + (nf * 16) * LS + kc * 32, LS, lane);
        a = wmma_bf16(qa[kc], bfr, a);
      }
      s[nf] = a;
    }

    // ---- scale + causal mask (only diagonal tile needs masking) ----
    const int qGlobBase = qt * 64 + w * 16 + half * 8;  // + r
    const int jGlobBase = jt * 64 + l16;                // + nf*16
#pragma unroll
    for (int nf = 0; nf < 4; ++nf)
#pragma unroll
      for (int r = 0; r < 8; ++r) {
        float v = s[nf][r] * scale;
        if (jt == qt && (jGlobBase + nf * 16) > (qGlobBase + r)) v = -1e30f;
        s[nf][r] = v;
      }

    // ---- online softmax: rows live across 16 lanes of each half-wave ----
#pragma unroll
    for (int r = 0; r < 8; ++r) {
      float mx = fmaxf(fmaxf(s[0][r], s[1][r]), fmaxf(s[2][r], s[3][r]));
#pragma unroll
      for (int off = 1; off < 16; off <<= 1)
        mx = fmaxf(mx, __shfl_xor(mx, off, 32));
      float mn = fmaxf(mloc[r], mx);
      float alpha = __expf(mloc[r] - mn);
      mloc[r] = mn;
      float psum = 0.0f;
#pragma unroll
      for (int nf = 0; nf < 4; ++nf) {
        float p = __expf(s[nf][r] - mn);
        s[nf][r] = p;
        psum += p;
      }
#pragma unroll
      for (int off = 1; off < 16; off <<= 1)
        psum += __shfl_xor(psum, off, 32);
      lloc[r] = lloc[r] * alpha + psum;
#pragma unroll
      for (int df = 0; df < 4; ++df) o_acc[df][r] *= alpha;
    }

    // ---- P -> per-wave LDS (convert to A-fragment layout via memory) ----
#pragma unroll
    for (int nf = 0; nf < 4; ++nf)
#pragma unroll
      for (int r = 0; r < 8; ++r)
        Pb[w][(half * 8 + r) * LS + nf * 16 + l16] = (bf16)s[nf][r];
    // (LDS ops from the same wave are in order: safe to read back below)

    // ---- O += P * V ----
#pragma unroll
    for (int df = 0; df < 4; ++df) {
#pragma unroll
      for (int kc = 0; kc < 2; ++kc) {
        v16bf pa = load_frag_a(&Pb[w][0] + kc * 32, LS, lane);
        v16bf vb = load_frag_b(Vt + (df * 16) * LS + kc * 32, LS, lane);
        o_acc[df] = wmma_bf16(pa, vb, o_acc[df]);
      }
    }
    __syncthreads();  // before next tile overwrites Ks/Vt
  }

  // ---- epilogue: O /= l, write ctx [b*T+t][h*64+d] ----
#pragma unroll
  for (int df = 0; df < 4; ++df)
#pragma unroll
    for (int r = 0; r < 8; ++r) {
      int qrow = qt * 64 + w * 16 + half * 8 + r;
      float v = o_acc[df][r] / lloc[r];
      O[(size_t)(b * T + qrow) * 2048 + h * 64 + df * 16 + l16] = (bf16)v;
    }
}

// ---------------------------------------------------------------------------
// Host-side launch
// ---------------------------------------------------------------------------
static inline size_t align256(size_t x) { return (x + 255) & ~(size_t)255; }

extern "C" void kernel_launch(void* const* d_in, const int* in_sizes, int n_in,
                              void* d_out, int out_size, void* d_ws, size_t ws_size,
                              hipStream_t stream) {
  (void)in_sizes; (void)n_in; (void)out_size; (void)ws_size;
  constexpr int B = 4, T = 2048, D = 2048, NKVD = 512;
  constexpr int M = B * T;  // 8192 rows

  const float* x  = (const float*)d_in[0];
  const float* Wq = (const float*)d_in[1];
  const float* Wk = (const float*)d_in[2];
  const float* Wv = (const float*)d_in[3];
  const float* Wo = (const float*)d_in[4];
  float* out = (float*)d_out;

  char* ws = (char*)d_ws;
  size_t off = 0;
  auto take = [&](size_t bytes) -> char* { char* p = ws + off; off += align256(bytes); return p; };

  bf16* xb  = (bf16*)take((size_t)M * D * 2);      // x in bf16
  bf16* Wqb = (bf16*)take((size_t)D * D * 2);
  bf16* Wkb = (bf16*)take((size_t)D * NKVD * 2);
  bf16* Wvb = (bf16*)take((size_t)D * NKVD * 2);
  bf16* Wob = (bf16*)take((size_t)D * D * 2);
  bf16* Qp  = (bf16*)take((size_t)M * D * 2);      // raw Q proj
  bf16* Kp  = (bf16*)take((size_t)M * NKVD * 2);   // raw K proj
  bf16* Vp  = (bf16*)take((size_t)M * NKVD * 2);   // V (used directly)
  bf16* Qr  = (bf16*)take((size_t)M * D * 2);      // RoPE'd Q
  bf16* Kr  = (bf16*)take((size_t)M * NKVD * 2);   // RoPE'd K
  bf16* ctx = (bf16*)take((size_t)M * D * 2);      // attention output

  // 1) fp32 -> bf16 conversions
  auto cvt = [&](const float* s, bf16* d, int n) {
    cvt_f32_bf16<<<(n + 255) / 256, 256, 0, stream>>>(s, d, n);
  };
  cvt(x,  xb,  M * D);
  cvt(Wq, Wqb, D * D);
  cvt(Wk, Wkb, D * NKVD);
  cvt(Wv, Wvb, D * NKVD);
  cvt(Wo, Wob, D * D);

  // 2) projections (WMMA GEMM, bf16 out)
  gemm_bf16_wmma<false><<<dim3(D / 128, M / 128), 256, 0, stream>>>(xb, Wqb, Qp, M, D, D);
  gemm_bf16_wmma<false><<<dim3(NKVD / 128, M / 128), 256, 0, stream>>>(xb, Wkb, Kp, M, NKVD, D);
  gemm_bf16_wmma<false><<<dim3(NKVD / 128, M / 128), 256, 0, stream>>>(xb, Wvb, Vp, M, NKVD, D);

  // 3) RoPE on Q and K
  {
    int qPairs = M * (D / 2);
    rope_bf16<<<(qPairs + 255) / 256, 256, 0, stream>>>(Qp, Qr, D, T, qPairs);
    int kPairs = M * (NKVD / 2);
    rope_bf16<<<(kPairs + 255) / 256, 256, 0, stream>>>(Kp, Kr, NKVD, T, kPairs);
  }

  // 4) causal flash attention (grid: q-tiles x heads x batch)
  flash_attn_wmma<<<dim3(T / 64, 32, B), 128, 0, stream>>>(Qr, Kr, Vp, ctx);

  // 5) output projection (fp32 out)
  gemm_bf16_wmma<true><<<dim3(D / 128, M / 128), 256, 0, stream>>>(ctx, Wob, out, M, D, D);
}
